// BayesianLinear_63084479643948
// MI455X (gfx1250) — matrix operand order repaired and checked by reading, compiled, not run on hardware
//
#include <hip/hip_runtime.h>

// ---------------------------------------------------------------------------
// BayesianLinear fused kernel for MI455X (gfx1250, wave32, WMMA).
//
//   W[t,m] = loc[m] + sum_n eps[t,n] * L[m,n]      (L lower triangular, N=2080)
//   y[t,k] = W[t,k] + sum_h x[t,h] * W[t, 32+32h+k]
//
// Strategy: f16 WMMA GEMM  E(16384x2080) x L^T, triangular K-bound per
// 32-wide m-chunk (halves FLOPs), epilogue folds W directly into y (32
// floats/token in LDS) so the 136MB W intermediate never touches memory.
// eps tile lives in LDS (64 tok x 2088 halves = 267KB, CDNA5-only), L lives
// in L2 (8.7MB f16, re-read by all 256 workgroups).
// ---------------------------------------------------------------------------

typedef __attribute__((ext_vector_type(16))) _Float16 v16h;
typedef __attribute__((ext_vector_type(8)))  _Float16 v8h;
typedef __attribute__((ext_vector_type(8)))  float    v8f;

#define N_TOT    2080           // IN_F*OUT_F + OUT_F
#define IN_FEAT  64
#define OUT_FEAT 32
#define LDK      2088           // padded row stride (halves) for L   (16B mult)
#define LDE      2088           // padded row stride (halves) for eps (dword stride 1044 -> conflict-free)
#define TOK_TILE 64
#define NCHUNK   65             // 2080 / 32
#define NTOK     16384          // B*S

union AFrag { v16h v; v8h h[2]; };

// --- Kernel 1: build L (row-major, f16, padded) in workspace ----------------
__global__ void build_L_f16(const float* __restrict__ logvar,
                            const float* __restrict__ cov,
                            _Float16* __restrict__ Lh) {
  int idx = blockIdx.x * blockDim.x + threadIdx.x;
  if (idx >= N_TOT * LDK) return;
  int m = idx / LDK;
  int n = idx - m * LDK;
  float v = 0.0f;
  if (n < m)       v = cov[(m * (m - 1)) / 2 + n];   // np.tril_indices(-1) order
  else if (n == m) v = __expf(0.5f * logvar[m]);
  Lh[idx] = (_Float16)v;
}

// one 16x16x32 f16 WMMA step; A from LDS eps tile, B from global L (L2-hot)
__device__ __forceinline__ v8f wmma_step(const _Float16* ap, const _Float16* bp, v8f acc) {
  AFrag a, b;
  a.h[0] = *(const v8h*)(ap);        // lanes<16: K0..7  | lanes>=16: K8..15
  a.h[1] = *(const v8h*)(ap + 16);   // lanes<16: K16..23| lanes>=16: K24..31
  b.h[0] = *(const v8h*)(bp);        // column n=lane%16, same K split as A
  b.h[1] = *(const v8h*)(bp + 16);
  return __builtin_amdgcn_wmma_f32_16x16x32_f16(false, a.v, false, b.v,
                                                (short)0, acc, false, false);
}

// --- Kernel 2: fused GEMM + epilogue ---------------------------------------
__global__ __launch_bounds__(256) void bayes_fused(
    const float* __restrict__ x,     // (NTOK, 64)
    const float* __restrict__ eps,   // (NTOK, 2080)
    const float* __restrict__ loc,   // (2080)
    const _Float16* __restrict__ Lh, // (2080, LDK) f16
    float* __restrict__ out)         // (NTOK, 32)
{
  extern __shared__ char smem[];
  _Float16* esh = (_Float16*)smem;                          // 64 x LDE halves
  float*    xsh = (float*)(smem + TOK_TILE * LDE * 2);      // 64 x 64
  float*    ysh = xsh + TOK_TILE * IN_FEAT;                 // 64 x 32

  const int tid = threadIdx.x;
  const int g0  = blockIdx.x * TOK_TILE;

  // ---- stage eps tile into LDS, f32 -> f16 (float4 loads, b64 LDS stores)
  for (int i = tid; i < TOK_TILE * (N_TOT / 4); i += 256) {
    int t  = i / (N_TOT / 4);
    int c4 = i - t * (N_TOT / 4);
    float4 v = ((const float4*)(eps + (size_t)(g0 + t) * N_TOT))[c4];
    _Float16* d = esh + t * LDE + c4 * 4;
    d[0] = (_Float16)v.x; d[1] = (_Float16)v.y;
    d[2] = (_Float16)v.z; d[3] = (_Float16)v.w;
  }
  // ---- stage x tile (kept f32 for the epilogue)
  for (int i = tid; i < TOK_TILE * IN_FEAT; i += 256) {
    int t = i >> 6, h = i & 63;
    xsh[i] = x[(size_t)(g0 + t) * IN_FEAT + h];
  }
  __syncthreads();

  // ---- y init: affine contribution of loc (W's "+loc" folded out of GEMM)
  for (int i = tid; i < TOK_TILE * OUT_FEAT; i += 256) {
    int t = i >> 5, k = i & 31;
    float acc = loc[k];
    const float* xr = xsh + t * IN_FEAT;
#pragma unroll 8
    for (int h = 0; h < IN_FEAT; ++h)
      acc += xr[h] * loc[OUT_FEAT + h * OUT_FEAT + k];
    ysh[i] = acc;
  }
  __syncthreads();

  const int lane = tid & 31;
  const int wave = tid >> 5;
  const int rb   = wave >> 1;      // token row-block 0..3 (16 tokens each)
  const int cb   = wave & 1;       // m col-block 0..1 (16 cols each)
  const int lh   = lane & 15;
  const int sel  = lane >> 4;

  const _Float16* arow = esh + (rb * 16 + lh) * LDE + sel * 8;
  const int t0 = rb * 16 + sel * 8;  // D layout: VGPR v holds token t0+v

  for (int c = 0; c < NCHUNK; ++c) {
    const int m_base = c * 32 + cb * 16;
    const _Float16* brow = Lh + (size_t)(m_base + lh) * LDK + sel * 8;

    // prefetch next chunk's B rows into cache (gfx1250 global_prefetch)
    if (c + 1 < NCHUNK)
      __builtin_prefetch(Lh + (size_t)(m_base + 32 + lh) * LDK + sel * 16, 0, 1);

    // triangular K bound: columns m in [32c, 32c+32) need n < 32(c+1)
    v8f acc0 = {};
    v8f acc1 = {};
    int ks = 0;
    for (; ks + 2 <= c + 1; ks += 2) {      // two interleaved acc chains
      acc0 = wmma_step(arow + ks * 32,       brow + ks * 32,       acc0);
      acc1 = wmma_step(arow + (ks + 1) * 32, brow + (ks + 1) * 32, acc1);
    }
    if (ks <= c)
      acc0 = wmma_step(arow + ks * 32, brow + ks * 32, acc0);

    // ---- epilogue: route W[t, m] into y[t, k].
    // Ownership of (t,k) is disjoint across waves: rb partitions t, cb
    // partitions k -> plain LDS read-modify-write is race-free.
    const int k = cb * 16 + lh;
    if (c == 0) {                 // m in [0,32): bias term, k == m
#pragma unroll
      for (int v = 0; v < 8; ++v)
        ysh[(t0 + v) * OUT_FEAT + k] += acc0[v] + acc1[v];
    } else {                      // m = 32 + 32h + k with h = c-1
      const int h = c - 1;
#pragma unroll
      for (int v = 0; v < 8; ++v) {
        int t = t0 + v;
        ysh[t * OUT_FEAT + k] += xsh[t * IN_FEAT + h] * (acc0[v] + acc1[v]);
      }
    }
  }
  __syncthreads();

  for (int i = tid; i < TOK_TILE * OUT_FEAT; i += 256)
    out[(size_t)g0 * OUT_FEAT + i] = ysh[i];
}

// ---------------------------------------------------------------------------
extern "C" void kernel_launch(void* const* d_in, const int* in_sizes, int n_in,
                              void* d_out, int out_size, void* d_ws, size_t ws_size,
                              hipStream_t stream) {
  (void)in_sizes; (void)n_in; (void)out_size; (void)ws_size;
  const float* x      = (const float*)d_in[0];
  const float* eps    = (const float*)d_in[1];
  const float* loc    = (const float*)d_in[2];
  const float* logvar = (const float*)d_in[3];
  const float* cov    = (const float*)d_in[4];
  float*    out = (float*)d_out;
  _Float16* Lh  = (_Float16*)d_ws;     // 2080*2088*2 = 8.69 MB workspace

  const int nL = N_TOT * LDK;
  build_L_f16<<<(nL + 255) / 256, 256, 0, stream>>>(logvar, cov, Lh);

  const size_t smem = (size_t)TOK_TILE * LDE * 2      // eps tile (f16)
                    + (size_t)TOK_TILE * IN_FEAT * 4  // x tile
                    + (size_t)TOK_TILE * OUT_FEAT * 4;// y accum  => 291,840 B
  bayes_fused<<<NTOK / TOK_TILE, 256, smem, stream>>>(x, eps, loc, Lh, out);
}
